// GNN_3547642986803
// MI455X (gfx1250) — matrix operand order repaired and checked by reading, not compiled, not run on hardware
//
#include <hip/hip_runtime.h>

// ---------------------------------------------------------------------------
// Problem constants (from the reference)
// ---------------------------------------------------------------------------
constexpr int T    = 32;
constexpr int NW   = 5;
constexpr int SUP  = 25;          // support nodes per task
constexpr int QRY  = 75;          // query nodes per task
constexpr int N    = 100;         // nodes per task
constexpr int EMB  = 256;
constexpr int C0   = 261;         // EMB + NW
constexpr int OUT  = 48;
constexpr int L    = 3;
constexpr int HID  = 128;
constexpr int HID2 = 192;
constexpr int TN   = T * N;       // 3200 node rows
constexpr int M    = T * N * N;   // 320000 edge rows
constexpr int XS   = 416;         // padded x row stride (>= 405)
constexpr int AS   = 384;         // padded agg row stride (>= 357)
constexpr int CFIN = C0 + L * OUT; // 405
constexpr float EPS = 1e-5f;

typedef _Float16     v16h __attribute__((ext_vector_type(16)));
typedef _Float16     v8h  __attribute__((ext_vector_type(8)));
typedef float        v8f  __attribute__((ext_vector_type(8)));
typedef unsigned int v4u  __attribute__((ext_vector_type(4)));
typedef int          v4i  __attribute__((ext_vector_type(4)));
typedef int          v8i  __attribute__((ext_vector_type(8)));

__device__ __forceinline__ v16h cat8(v8h lo, v8h hi) {
  return __builtin_shufflevector(lo, hi, 0,1,2,3,4,5,6,7,8,9,10,11,12,13,14,15);
}

// ---------------------------------------------------------------------------
// Initial node features: x[:, :256] = embedding, x[:, 256:261] = support one-hot
// (query rows zero), padding zeroed.
// ---------------------------------------------------------------------------
__global__ void init_x_kernel(const float* __restrict__ emb,
                              const float* __restrict__ supy,
                              float* __restrict__ x) {
  int idx = blockIdx.x * blockDim.x + threadIdx.x;
  if (idx >= TN * XS) return;
  int ti = idx / XS, c = idx - ti * XS;
  int t = ti / N, n = ti - t * N;
  float v = 0.f;
  if (c < EMB)                 v = emb[(size_t)ti * EMB + c];
  else if (c < C0 && n < SUP)  v = supy[(t * SUP + n) * NW + (c - EMB)];
  x[idx] = v;
}

// ---------------------------------------------------------------------------
// Per-stage weight pre-conversion to f16 (so the GEMM B-tile is a byte-exact
// 2D tile copy, eligible for the Tensor Data Mover).
// ---------------------------------------------------------------------------
__global__ void w2h_kernel(const float* __restrict__ W, _Float16* __restrict__ W16, int n) {
  int i = blockIdx.x * blockDim.x + threadIdx.x;
  if (i < n) W16[i] = (_Float16)W[i];
}

// ---------------------------------------------------------------------------
// WMMA edge-MLP GEMM stage with fused BN-stat accumulation.
//   MODE 0: A[m,k] = |x[t,i,k] - x[t,j,k]|  (m = (t*N+i)*N + j), K padded to 32
//   MODE 1: A[m,k] = lrelu(alpha[k]*h_in[m,k] + beta[k])
//   Output: h_out[m, :COUT] (f16) + per-channel sum / sumsq (f32 atomics).
// Block: 256 threads = 8 waves as 4 (row tiles of 16) x 2 (col halves).
// CT = COUT/32 output tiles of 16x16 per wave; v_wmma_f32_16x16x32_f16.
// B tile (32 x COUT f16 weights, transposed layout Bt[c][k]) is streamed into
// LDS by the TDM (tensor_load_to_lds) from wave 0; TDM OOB reads return zero,
// which supplies the K-padding for the un-even first-stage K (261/309/357).
// ---------------------------------------------------------------------------
template <int CT, int MODE>
__global__ __launch_bounds__(256)
void gemm_bn_kernel(const _Float16* __restrict__ hin,
                    const float*    __restrict__ xg,
                    const _Float16* __restrict__ W16,    // [COUT, Kc] row-major f16
                    const float*    __restrict__ alpha,
                    const float*    __restrict__ beta,
                    _Float16*       __restrict__ hout,   // [M, COUT]
                    float* __restrict__ osum, float* __restrict__ osq,
                    int Kc, int Kin) {
  constexpr int COUT = CT * 32;
  __shared__ __align__(16) _Float16 At[64 * 32];
  __shared__ __align__(16) _Float16 Bt[192 * 32];   // transposed: Bt[c][k]
  __shared__ float sb[192];
  __shared__ float qb[192];
  __shared__ int   rI[64];
  __shared__ int   rJ[64];

  const int tid   = threadIdx.x;
  const int lane  = tid & 31;
  const int wave  = tid >> 5;
  const int waveR = wave & 3;
  const int waveC = wave >> 2;
  const int rowbase = blockIdx.x * 64 + waveR * 16;
  const int colbase = waveC * (COUT / 2);

  for (int i = tid; i < 192; i += 256) { sb[i] = 0.f; qb[i] = 0.f; }
  if (MODE == 0 && tid < 64) {
    // hoist the m -> (t,i,j) decomposition out of the K loop (K-invariant)
    int m   = blockIdx.x * 64 + tid;
    int t   = m / (N * N);
    int rem = m - t * (N * N);
    int ii  = rem / N;
    int jj  = rem - ii * N;
    rI[tid] = (t * N + ii) * XS;
    rJ[tid] = (t * N + jj) * XS;
  }

  v8f acc[CT] = {};

  const int kIters = (Kc + 31) >> 5;
  for (int kt = 0; kt < kIters; ++kt) {
    __syncthreads();
    // ---- stage A tile (64 rows x 32 k), computed on the fly, cvt f16 ----
    for (int e = tid; e < 64 * 32; e += 256) {
      int r = e >> 5, k = e & 31;
      int kg = kt * 32 + k;
      float v = 0.f;
      if (MODE == 0) {
        if (kg < Kc) v = fabsf(xg[rI[r] + kg] - xg[rJ[r] + kg]);
      } else {
        int m = blockIdx.x * 64 + r;
        float h = (float)hin[(size_t)m * Kin + kg];
        float a = fmaf(alpha[kg], h, beta[kg]);
        v = (a >= 0.f) ? a : 0.2f * a;
      }
      At[e] = (_Float16)v;
    }
    // ---- stage B tile transposed: Bt[c][k] = W16[c][kt*32+k] ----
#if __has_builtin(__builtin_amdgcn_tensor_load_to_lds) && __has_builtin(__builtin_amdgcn_s_wait_tensorcnt)
    if (wave == 0) {
      // Tensor DMA descriptor (D#): 2D tile, dim0 = 32 f16 (contiguous K),
      // dim1 = COUT rows with row stride Kc; OOB (kg >= Kc) reads return 0.
      unsigned long long gaddr =
          (unsigned long long)(size_t)(W16 + (size_t)kt * 32);         // byte addr of tile start
      unsigned int ldsa = (unsigned int)(size_t)(&Bt[0]);              // LDS byte offset (addr[31:0])
      int remaining = Kc - kt * 32;                                    // tensor_dim0 from tile start
      v4u g0;
      g0[0] = 1u;                                                      // count=1 (valid user D#)
      g0[1] = ldsa;                                                    // lds_addr
      g0[2] = (unsigned int)(gaddr & 0xffffffffu);                     // global_addr[31:0]
      g0[3] = (unsigned int)((gaddr >> 32) & 0x01ffffffu)              // global_addr[56:32]
            | 0x80000000u;                                             // type=2 ("image")
      v8i g1;
      g1[0] = 0x10000;                                                 // data_size=1 (2 bytes)
      g1[1] = (int)(((unsigned)remaining & 0xffffu) << 16);            // tensor_dim0[15:0]
      g1[2] = (int)(((unsigned)COUT & 0xffffu) << 16);                 // dim0 hi=0 | tensor_dim1 lo
      g1[3] = (int)(32u << 16);                                        // dim1 hi=0 | tile_dim0=32
      g1[4] = COUT;                                                    // tile_dim1=COUT, tile_dim2=0
      g1[5] = Kc;                                                      // tensor_dim0_stride[31:0]
      g1[6] = 0;                                                       // stride hi | dim1_stride lo
      g1[7] = 0;
      v4i g2 = {0, 0, 0, 0};                                           // 2D tensor: groups 2/3 unused
      v4i g3 = {0, 0, 0, 0};
      v8i g4 = {0, 0, 0, 0, 0, 0, 0, 0};                               // 6-arg form extra group
      __builtin_amdgcn_tensor_load_to_lds(g0, g1, g2, g3, g4, 0);
      __builtin_amdgcn_s_wait_tensorcnt(0);
    }
#else
    for (int e = tid; e < COUT * 32; e += 256) {
      int c = e >> 5, k = e & 31;
      int kg = kt * 32 + k;
      Bt[e] = (kg < Kc) ? W16[c * Kc + kg] : (_Float16)0.f;
    }
#endif
    if (MODE == 1 && kt + 1 < kIters) {
      // gfx1250 global_prefetch_b8 on the next A stripe
      __builtin_prefetch(hin + (size_t)(blockIdx.x * 64 + (tid >> 2)) * Kin + (kt + 1) * 32, 0, 1);
    }
    __syncthreads();

    // ---- fragments per CDNA5 16-bit WMMA lane layouts ----
    // A 16x32: lane L(0..15)=row L, halves 0..7 -> K=0..7, 8..15 -> K=16..23;
    //          lane L+16: same row, K=8..15 and K=24..31.
    const int lrow = lane & 15;
    const int koff = (lane >> 4) << 3;
    const _Float16* ar = &At[lrow * 32];
    v16h afrag = cat8(*(const v8h*)(ar + koff), *(const v8h*)(ar + 16 + koff));
    // B 32x16: lane n(0..15)=col n, K=0..7,16..23 ; lane n+16: col n, K=8..15,24..31.
    const int kb = (lane >> 4) << 3;
#pragma unroll
    for (int ct = 0; ct < CT; ++ct) {
      const _Float16* br = &Bt[(colbase + ct * 16 + lrow) * 32 + kb];
      v16h bfrag = cat8(*(const v8h*)br, *(const v8h*)(br + 16));
      acc[ct] = __builtin_amdgcn_wmma_f32_16x16x32_f16(
          false, afrag, false, bfrag, (short)0, acc[ct], false, false);
    }
  }

  // ---- epilogue: store f16 h, accumulate per-channel sum/sumsq in f32 ----
  // C/D layout: VGPR r, lanes 0..15 -> row r, lanes 16..31 -> row r+8; col = lane&15.
  const int rb = rowbase + ((lane >> 4) << 3);
  const int cl = lane & 15;
#pragma unroll
  for (int ct = 0; ct < CT; ++ct) {
    int col = colbase + ct * 16 + cl;
    float ls = 0.f, lq = 0.f;
#pragma unroll
    for (int r = 0; r < 8; ++r) {
      float v = acc[ct][r];
      hout[(size_t)(rb + r) * COUT + col] = (_Float16)v;
      ls += v;
      lq += v * v;
    }
    atomicAdd(&sb[col], ls);
    atomicAdd(&qb[col], lq);
  }
  __syncthreads();
  for (int c = tid; c < COUT; c += 256) {
    atomicAdd(&osum[c], sb[c]);
    atomicAdd(&osq[c], qb[c]);
  }
}

// ---------------------------------------------------------------------------
// Fold BN stats + gamma/beta into per-channel affine (alpha, beta).
// ---------------------------------------------------------------------------
__global__ void bn_finalize_kernel(const float* __restrict__ s,
                                   const float* __restrict__ q,
                                   const float* __restrict__ g,
                                   const float* __restrict__ b,
                                   float* __restrict__ alpha,
                                   float* __restrict__ beta,
                                   int C, float invM) {
  int c = blockIdx.x * blockDim.x + threadIdx.x;
  if (c >= C) return;
  float m  = s[c] * invM;
  float v  = q[c] * invM - m * m;
  float rs = rsqrtf(v + EPS);
  float a  = g[c] * rs;
  alpha[c] = a;
  beta[c]  = b[c] - m * a;
}

// ---------------------------------------------------------------------------
// Stage 5: s[m] = dot(lrelu(bn(h4[m,:])), w5) + bias5 ; stats of s.
// One wave per row (coalesced f16 loads), shfl reduction.
// ---------------------------------------------------------------------------
__global__ __launch_bounds__(256)
void h5_kernel(const _Float16* __restrict__ h4,
               const float* __restrict__ alpha, const float* __restrict__ beta,
               const float* __restrict__ w5, const float* __restrict__ bias5,
               float* __restrict__ sout,
               float* __restrict__ osum, float* __restrict__ osq) {
  const int lane = threadIdx.x & 31;
  const int wave = threadIdx.x >> 5;
  const int m = blockIdx.x * 8 + wave;
  float acc = 0.f;
  for (int k = lane; k < HID; k += 32) {
    float h = (float)h4[(size_t)m * HID + k];
    float a = fmaf(alpha[k], h, beta[k]);
    a = (a >= 0.f) ? a : 0.2f * a;
    acc += a * w5[k];
  }
#pragma unroll
  for (int off = 16; off > 0; off >>= 1) acc += __shfl_xor(acc, off, 32);
  __shared__ float bs, bq;
  if (threadIdx.x == 0) { bs = 0.f; bq = 0.f; }
  __syncthreads();
  if (lane == 0) {
    float sv = acc + bias5[0];
    sout[m] = sv;
    atomicAdd(&bs, sv);
    atomicAdd(&bq, sv * sv);
  }
  __syncthreads();
  if (threadIdx.x == 0) { atomicAdd(osum, bs); atomicAdd(osq, bq); }
}

// ---------------------------------------------------------------------------
// Stage 6: per (t,i): w[j] = softmax_j( bn(s) - 1e8*[j==i] )
// ---------------------------------------------------------------------------
__global__ void softmax_kernel(const float* __restrict__ s,
                               const float* __restrict__ alpha,
                               const float* __restrict__ beta,
                               float* __restrict__ wout) {
  int ti = blockIdx.x;              // 0..TN-1
  int i  = ti % N;
  float a = alpha[0], b = beta[0];
  __shared__ float buf[128];
  int j = threadIdx.x;              // 128 threads, j < N valid
  float v = -3.0e38f;
  if (j < N) {
    v = fmaf(a, s[ti * N + j], b);
    if (j == i) v -= 1e8f;
  }
  buf[j] = v;
  __syncthreads();
  for (int o = 64; o > 0; o >>= 1) {
    if (j < o) buf[j] = fmaxf(buf[j], buf[j + o]);
    __syncthreads();
  }
  float mx = buf[0];
  __syncthreads();
  float e = (j < N) ? __expf(v - mx) : 0.f;
  buf[j] = e;
  __syncthreads();
  for (int o = 64; o > 0; o >>= 1) {
    if (j < o) buf[j] += buf[j + o];
    __syncthreads();
  }
  float inv = 1.f / buf[0];
  if (j < N) wout[ti * N + j] = e * inv;
}

// ---------------------------------------------------------------------------
// Stage 7a: agg[t,i,c] = sum_j w[t,i,j] * x[t,j,c]
// ---------------------------------------------------------------------------
__global__ __launch_bounds__(256)
void agg_kernel(const float* __restrict__ w, const float* __restrict__ x,
                float* __restrict__ agg, int Kc) {
  int ti = blockIdx.x;
  int t  = ti / N;
  __shared__ float wl[128];
  if (threadIdx.x < N) wl[threadIdx.x] = w[ti * N + threadIdx.x];
  __syncthreads();
  for (int c = threadIdx.x; c < Kc; c += 256) {
    float acc = 0.f;
    for (int j = 0; j < N; ++j) acc += wl[j] * x[(t * N + j) * XS + c];
    agg[ti * AS + c] = acc;
  }
}

// ---------------------------------------------------------------------------
// Stage 7b: nn[row,o] = [agg, x] . lw[o,:] + lb[o], stats over TN rows.
// ---------------------------------------------------------------------------
__global__ void nn_kernel(const float* __restrict__ agg,
                          const float* __restrict__ x,
                          const float* __restrict__ lw,   // [OUT, 2*Kc]
                          const float* __restrict__ lb,
                          float* __restrict__ nn,
                          float* __restrict__ osum, float* __restrict__ osq,
                          int Kc) {
  int ti = blockIdx.x;
  int o  = threadIdx.x;
  if (o >= OUT) return;
  const float* lwr = lw + (size_t)o * 2 * Kc;
  const float* ar  = agg + ti * AS;
  const float* xr  = x + ti * XS;
  float acc = lb[o];
  for (int c = 0; c < Kc; ++c) acc += ar[c] * lwr[c];
  for (int c = 0; c < Kc; ++c) acc += xr[c] * lwr[Kc + c];
  nn[ti * OUT + o] = acc;
  atomicAdd(&osum[o], acc);
  atomicAdd(&osq[o], acc * acc);
}

// ---------------------------------------------------------------------------
// Stage 8: x_new = concat( bn(nn), x_old[:Kc] ), padding zeroed.
// ---------------------------------------------------------------------------
__global__ void xnew_kernel(const float* __restrict__ nn,
                            const float* __restrict__ xold,
                            const float* __restrict__ alpha,
                            const float* __restrict__ beta,
                            float* __restrict__ xnew, int Kc) {
  int idx = blockIdx.x * blockDim.x + threadIdx.x;
  if (idx >= TN * XS) return;
  int ti = idx / XS, c = idx - ti * XS;
  float v = 0.f;
  if (c < OUT)            v = fmaf(alpha[c], nn[ti * OUT + c], beta[c]);
  else if (c < OUT + Kc)  v = xold[ti * XS + (c - OUT)];
  xnew[idx] = v;
}

// ---------------------------------------------------------------------------
// Final: logits = x @ fc_w^T + fc_b ; loss (all rows) + acc (query rows).
// ---------------------------------------------------------------------------
__global__ void loss_kernel(const float* __restrict__ x,
                            const float* __restrict__ fcw,   // [NW, CFIN]
                            const float* __restrict__ fcb,
                            const float* __restrict__ supy,
                            const float* __restrict__ qryy,
                            float* __restrict__ acc2) {
  int r = blockIdx.x * blockDim.x + threadIdx.x;
  if (r >= TN) return;
  int t = r / N, n = r - t * N;
  const float* xr = x + (size_t)r * XS;
  float lg[NW];
  for (int w = 0; w < NW; ++w) {
    const float* wr = fcw + (size_t)w * CFIN;
    float a = fcb[w];
    for (int c = 0; c < CFIN; ++c) a += xr[c] * wr[c];
    lg[w] = a;
  }
  const float* yrow = (n < SUP) ? (supy + (t * SUP + n) * NW)
                                : (qryy + (t * QRY + (n - SUP)) * NW);
  int gt = 0; float best = yrow[0];
  for (int w = 1; w < NW; ++w) if (yrow[w] > best) { best = yrow[w]; gt = w; }
  float mx = lg[0];
  for (int w = 1; w < NW; ++w) mx = fmaxf(mx, lg[w]);
  float se = 0.f;
  for (int w = 0; w < NW; ++w) se += __expf(lg[w] - mx);
  float logp = lg[gt] - mx - __logf(se);
  atomicAdd(&acc2[0], -logp);
  if (n >= SUP) {
    int pm = 0; float pb = lg[0];
    for (int w = 1; w < NW; ++w) if (lg[w] > pb) { pb = lg[w]; pm = w; }
    atomicAdd(&acc2[1], (pm == gt) ? 1.f : 0.f);
  }
}

__global__ void writeout_kernel(const float* __restrict__ acc2, float* __restrict__ out) {
  if (threadIdx.x == 0) {
    out[0] = acc2[0] / (float)TN;
    out[1] = acc2[1] / (float)(T * QRY);
  }
}

// ---------------------------------------------------------------------------
// Host orchestration. Input order: setup_inputs() insertion order:
//   [0] embedding, [1] support_y, [2] query_y, then `params` flattened as a
//   JAX pytree (dict keys sorted): [3] fc_b, [4] fc_w, then layers 0..2 with
//   per-layer sorted keys:
//   b1 b2 b3 b4 b5 bias5 g1 g2 g3 g4 g5 lb lbe lg lw w1 w2 w3 w4 w5
// ---------------------------------------------------------------------------
extern "C" void kernel_launch(void* const* d_in, const int* in_sizes, int n_in,
                              void* d_out, int out_size, void* d_ws, size_t ws_size,
                              hipStream_t stream) {
  (void)in_sizes; (void)n_in; (void)out_size;
  const float* emb  = (const float*)d_in[0];
  const float* supy = (const float*)d_in[1];
  const float* qryy = (const float*)d_in[2];
  const float* fcb  = (const float*)d_in[3];
  const float* fcw  = (const float*)d_in[4];
  auto LP = [&](int l, int k) -> const float* { return (const float*)d_in[5 + l * 20 + k]; };
  enum { P_B1=0, P_B2, P_B3, P_B4, P_B5, P_BIAS5, P_G1, P_G2, P_G3, P_G4, P_G5,
         P_LB, P_LBE, P_LG, P_LW, P_W1, P_W2, P_W3, P_W4, P_W5 };

  // --- workspace carve-out ---
  char*  base = (char*)d_ws;
  size_t off  = 0;
  auto alloc = [&](size_t bytes) -> void* {
    off = (off + 255) & ~(size_t)255;
    void* p = base + off;
    off += bytes;
    return p;
  };
  _Float16* hA   = (_Float16*)alloc((size_t)M * 192 * sizeof(_Float16));
  _Float16* hB   = (_Float16*)alloc((size_t)M * 192 * sizeof(_Float16));
  _Float16* w16  = (_Float16*)alloc((size_t)192 * 384 * sizeof(_Float16)); // stage weights, f16
  float* xA      = (float*)alloc((size_t)TN * XS * sizeof(float));
  float* xB      = (float*)alloc((size_t)TN * XS * sizeof(float));
  float* sbuf    = (float*)alloc((size_t)M * sizeof(float));
  float* wbuf    = (float*)alloc((size_t)M * sizeof(float));
  float* aggbuf  = (float*)alloc((size_t)TN * AS * sizeof(float));
  float* nnbuf   = (float*)alloc((size_t)TN * OUT * sizeof(float));
  float* stats   = (float*)alloc(768 * sizeof(float));
  float* lacc    = (float*)alloc(2 * sizeof(float));
  (void)ws_size;

  float* ssum = stats;        // [192]
  float* ssq  = stats + 192;  // [192]
  float* sal  = stats + 384;  // [192] alpha
  float* sbe  = stats + 576;  // [192] beta
  const float invM  = 1.f / (float)M;
  const float invTN = 1.f / (float)TN;

  auto cvtW = [&](const float* W, int n) {
    w2h_kernel<<<(n + 255) / 256, 256, 0, stream>>>(W, w16, n);
  };

  init_x_kernel<<<(TN * XS + 255) / 256, 256, 0, stream>>>(emb, supy, xA);

  float* xcur = xA;
  float* xnx  = xB;
  for (int l = 0; l < L; ++l) {
    const int c = C0 + OUT * l;   // 261, 309, 357

    // S1: |xi - xj| -> 128ch  (pairwise edge GEMM, WMMA)
    (void)hipMemsetAsync(stats, 0, 384 * sizeof(float), stream);
    cvtW(LP(l, P_W1), 128 * c);
    gemm_bn_kernel<4, 0><<<M / 64, 256, 0, stream>>>(
        hA, xcur, w16, sal, sbe, hA, ssum, ssq, c, 1);
    bn_finalize_kernel<<<2, 64, 0, stream>>>(ssum, ssq, LP(l, P_G1), LP(l, P_B1), sal, sbe, 128, invM);

    // S2: 128 -> 192
    (void)hipMemsetAsync(stats, 0, 384 * sizeof(float), stream);
    cvtW(LP(l, P_W2), 192 * 128);
    gemm_bn_kernel<6, 1><<<M / 64, 256, 0, stream>>>(
        hA, xcur, w16, sal, sbe, hB, ssum, ssq, HID, HID);
    bn_finalize_kernel<<<3, 64, 0, stream>>>(ssum, ssq, LP(l, P_G2), LP(l, P_B2), sal, sbe, 192, invM);

    // S3: 192 -> 192
    (void)hipMemsetAsync(stats, 0, 384 * sizeof(float), stream);
    cvtW(LP(l, P_W3), 192 * 192);
    gemm_bn_kernel<6, 1><<<M / 64, 256, 0, stream>>>(
        hB, xcur, w16, sal, sbe, hA, ssum, ssq, HID2, HID2);
    bn_finalize_kernel<<<3, 64, 0, stream>>>(ssum, ssq, LP(l, P_G3), LP(l, P_B3), sal, sbe, 192, invM);

    // S4: 192 -> 128
    (void)hipMemsetAsync(stats, 0, 384 * sizeof(float), stream);
    cvtW(LP(l, P_W4), 128 * 192);
    gemm_bn_kernel<4, 1><<<M / 64, 256, 0, stream>>>(
        hA, xcur, w16, sal, sbe, hB, ssum, ssq, HID2, HID2);
    bn_finalize_kernel<<<2, 64, 0, stream>>>(ssum, ssq, LP(l, P_G4), LP(l, P_B4), sal, sbe, 128, invM);

    // S5: 128 -> scalar edge score
    (void)hipMemsetAsync(stats, 0, 384 * sizeof(float), stream);
    h5_kernel<<<M / 8, 256, 0, stream>>>(hB, sal, sbe, LP(l, P_W5), LP(l, P_BIAS5), sbuf, ssum, ssq);
    bn_finalize_kernel<<<1, 64, 0, stream>>>(ssum, ssq, LP(l, P_G5), LP(l, P_B5), sal, sbe, 1, invM);

    // S6: softmax over neighbors (diagonal masked)
    softmax_kernel<<<TN, 128, 0, stream>>>(sbuf, sal, sbe, wbuf);

    // S7: aggregate + node linear (+ BN stats over TN rows)
    agg_kernel<<<TN, 256, 0, stream>>>(wbuf, xcur, aggbuf, c);
    (void)hipMemsetAsync(stats, 0, 384 * sizeof(float), stream);
    nn_kernel<<<TN, 64, 0, stream>>>(aggbuf, xcur, LP(l, P_LW), LP(l, P_LB), nnbuf, ssum, ssq, c);
    bn_finalize_kernel<<<1, 64, 0, stream>>>(ssum, ssq, LP(l, P_LG), LP(l, P_LBE), sal, sbe, OUT, invTN);

    // S8: x <- concat(bn(nn), x)
    xnew_kernel<<<(TN * XS + 255) / 256, 256, 0, stream>>>(nnbuf, xcur, sal, sbe, xnx, c);
    float* tmp = xcur; xcur = xnx; xnx = tmp;
  }

  // Final FC + loss/accuracy
  (void)hipMemsetAsync(lacc, 0, 2 * sizeof(float), stream);
  loss_kernel<<<(TN + 255) / 256, 256, 0, stream>>>(xcur, fcw, fcb, supy, qryy, lacc);
  writeout_kernel<<<1, 32, 0, stream>>>(lacc, (float*)d_out);
}
